// Maj3FC_463856468333
// MI455X (gfx1250) — 8-bit matrix operand order repaired and checked by reading, compiled, not run on hardware
//
#include <hip/hip_runtime.h>
#include <stdint.h>


typedef int v8i __attribute__((ext_vector_type(8)));

#define MDIM   512
#define NDIM   512
#define CIN    1536
#define GROUPS 512
#define KDIM   6656
#define KT_CNT 104
#define RT_CNT 32
#define FRAG_BYTES 1024

__device__ __forceinline__ int fsgn(float v) { return (v > 0.0f) - (v < 0.0f); }

__device__ __forceinline__ int featval(const float* __restrict__ src, int row,
                                       int k, bool wside) {
    const float* rp = src + (size_t)row * CIN;
    if (k < CIN) {
        int s = fsgn(rp[k]);
        return wside ? (s << 1) : s;
    }
    int e = k - CIN;
    int g = e / 10;
    int t = e - g * 10;
    float f0 = rp[3 * g + 0], f1 = rp[3 * g + 1], f2 = rp[3 * g + 2];
    int s0 = fsgn(f0), s1 = fsgn(f1), s2 = fsgn(f2);
    int m0 = s0 * s0, m1 = s1 * s1, m2 = s2 * s2;
    int val, coef;
    switch (t) {
        case 0: val = s0 * m1;        coef = -1; break;
        case 1: val = s0 * m2;        coef = -1; break;
        case 2: val = s1 * m0;        coef = -1; break;
        case 3: val = s1 * m2;        coef = -1; break;
        case 4: val = s2 * m0;        coef = -1; break;
        case 5: val = s2 * m1;        coef = -1; break;
        case 6: val = s0 * m1 * m2;   coef =  1; break;
        case 7: val = s1 * m0 * m2;   coef =  1; break;
        case 8: val = s2 * m0 * m1;   coef =  1; break;
        default: val = s0 * s1 * s2;  coef = -1; break;
    }
    return wside ? coef * val : val;
}

template <bool BSIDE>
__global__ void __launch_bounds__(256)
maj3_pack(const float* __restrict__ src, int8_t* __restrict__ dst) {
    int t    = blockIdx.x * blockDim.x + threadIdx.x;
    int lane = t & 31;
    int frag = t >> 5;
    if (frag >= RT_CNT * KT_CNT) return;
    int rt = frag / KT_CNT;
    int kt = frag - rt * KT_CNT;
    int row = rt * 16 + (lane & 15);
    int hi  = (lane >= 16);

    int words[8];
    for (int v = 0; v < 8; ++v) {
        int w32 = 0;
        for (int j = 0; j < 4; ++j) {
            int kl;
            if (BSIDE) {
                kl = ((v >> 1) * 16) + (v & 1) * 4 + j + (hi ? 8 : 0);
            } else {
                kl = ((v >> 1) * 16) + (v & 1) * 4 + j + (hi ? 8 : 0);
            }
            int f = featval(src, row, kt * 64 + kl, BSIDE);
            w32 |= (f & 0xff) << (8 * j);
        }
        words[v] = w32;
    }
    int4* out4 = (int4*)(dst + ((size_t)frag * 32 + lane) * 32);
    out4[0] = make_int4(words[0], words[1], words[2], words[3]);
    out4[1] = make_int4(words[4], words[5], words[6], words[7]);
}

__global__ void __launch_bounds__(128)
maj3_wmma_gemm(const int8_t* __restrict__ Apack, const int8_t* __restrict__ Bpack,
               float* __restrict__ out) {
    const int lane = threadIdx.x & 31;
    const int wid  = blockIdx.x * (blockDim.x >> 5) + (threadIdx.x >> 5);
    const int tm = wid >> 4;
    const int tn = wid & 15;

    const int8_t* a0p = Apack + (size_t)(tm * 2 + 0) * KT_CNT * FRAG_BYTES + lane * 32;
    const int8_t* a1p = Apack + (size_t)(tm * 2 + 1) * KT_CNT * FRAG_BYTES + lane * 32;
    const int8_t* b0p = Bpack + (size_t)(tn * 2 + 0) * KT_CNT * FRAG_BYTES + lane * 32;
    const int8_t* b1p = Bpack + (size_t)(tn * 2 + 1) * KT_CNT * FRAG_BYTES + lane * 32;

    v8i acc00 = {0,0,0,0,0,0,0,0};
    v8i acc01 = {0,0,0,0,0,0,0,0};
    v8i acc10 = {0,0,0,0,0,0,0,0};
    v8i acc11 = {0,0,0,0,0,0,0,0};

    for (int kt = 0; kt < KT_CNT; ++kt) {
        v8i A0 = *(const v8i*)a0p;
        v8i A1 = *(const v8i*)a1p;
        v8i B0 = *(const v8i*)b0p;
        v8i B1 = *(const v8i*)b1p;
        a0p += FRAG_BYTES; a1p += FRAG_BYTES;
        b0p += FRAG_BYTES; b1p += FRAG_BYTES;
        acc00 = __builtin_amdgcn_wmma_i32_16x16x64_iu8(true, A0, true, B0, acc00, false, false);
        acc01 = __builtin_amdgcn_wmma_i32_16x16x64_iu8(true, A0, true, B1, acc01, false, false);
        acc10 = __builtin_amdgcn_wmma_i32_16x16x64_iu8(true, A1, true, B0, acc10, false, false);
        acc11 = __builtin_amdgcn_wmma_i32_16x16x64_iu8(true, A1, true, B1, acc11, false, false);
    }

    const int nlo  = lane & 15;
    const int mofs = (lane < 16) ? 0 : 8;
    const int m0 = tm * 32, n0 = tn * 32;
    for (int v = 0; v < 8; ++v) {
        int mA = m0 + mofs + v;
        int mB = m0 + 16 + mofs + v;
        out[(size_t)mA * NDIM + n0 + nlo]      = 1.125f * (float)acc00[v];
        out[(size_t)mA * NDIM + n0 + 16 + nlo] = 1.125f * (float)acc01[v];
        out[(size_t)mB * NDIM + n0 + nlo]      = 1.125f * (float)acc10[v];
        out[(size_t)mB * NDIM + n0 + 16 + nlo] = 1.125f * (float)acc11[v];
    }
}

__global__ void __launch_bounds__(256)
maj3_fallback(const float* __restrict__ x, const float* __restrict__ w,
              float* __restrict__ out) {
    int idx = blockIdx.x * blockDim.x + threadIdx.x;
    if (idx >= MDIM * NDIM) return;
    int b = idx / NDIM, c = idx - (idx / NDIM) * NDIM;
    const float* xp = x + (size_t)b * CIN;
    const float* wp = w + (size_t)c * CIN;
    int total = 0;
    for (int g = 0; g < GROUPS; ++g) {
        int s = fsgn(xp[3*g+0]) * fsgn(wp[3*g+0])
              + fsgn(xp[3*g+1]) * fsgn(wp[3*g+1])
              + fsgn(xp[3*g+2]) * fsgn(wp[3*g+2]);
        total += (s > 1) ? 1 : ((s < -1) ? -1 : s);
    }
    out[idx] = 2.25f * (float)total;
}

extern "C" void kernel_launch(void* const* d_in, const int* in_sizes, int n_in,
                              void* d_out, int out_size, void* d_ws, size_t ws_size,
                              hipStream_t stream) {
    const float* x = (const float*)d_in[0];
    const float* w = (const float*)d_in[1];
    float* out = (float*)d_out;

    const size_t packBytes = (size_t)MDIM * KDIM;
    if (ws_size < 2 * packBytes) {
        maj3_fallback<<<(MDIM * NDIM + 255) / 256, 256, 0, stream>>>(x, w, out);
        return;
    }
    int8_t* Apack = (int8_t*)d_ws;
    int8_t* Bpack = Apack + packBytes;

    const int packThreads = RT_CNT * KT_CNT * 32;
    maj3_pack<false><<<(packThreads + 255) / 256, 256, 0, stream>>>(x, Apack);
    maj3_pack<true ><<<(packThreads + 255) / 256, 256, 0, stream>>>(w, Bpack);

    maj3_wmma_gemm<<<64, 128, 0, stream>>>(Apack, Bpack, out);
}
